// DeepseekV3MoE_24902220382975
// MI455X (gfx1250) — matrix operand order repaired and checked by reading, not compiled, not run on hardware
//
#include <hip/hip_runtime.h>
#include <math.h>

// ---------------- problem constants (match reference) ----------------
#define T_TOK 1024
#define H_DIM 1024
#define I_DIM 512
#define E_EXP 64
#define S_SH  2
#define G_GRP 8
#define TOPK_GROUP 4
#define TOPK 8

typedef __attribute__((ext_vector_type(16))) _Float16 v16h;
typedef __attribute__((ext_vector_type(8)))  float    v8f;

// ---------------- gating: grouped top-k router ----------------
// grid = T blocks, 64 threads (one per expert). Thread 0 does the serial
// grouped top-k (64 candidates -> 8 selections), mirroring jax.lax.top_k
// tie-breaking (first/lowest index wins on ties).
__global__ void moe_gate_kernel(const float* __restrict__ x,
                                const float* __restrict__ gate_w,
                                int* __restrict__ counts,
                                int* __restrict__ tok_list,
                                float* __restrict__ wt_list)
{
  __shared__ float xrow[H_DIM];
  __shared__ float logits[E_EXP];
  const int t   = blockIdx.x;
  const int tid = threadIdx.x;

  for (int c = tid; c < H_DIM; c += 64) xrow[c] = x[(size_t)t * H_DIM + c];
  __syncthreads();

  {
    const float* w = gate_w + (size_t)tid * H_DIM;
    float acc = 0.f;
#pragma unroll 8
    for (int i = 0; i < H_DIM; ++i) acc += xrow[i] * w[i];
    logits[tid] = acc;
  }
  __syncthreads();

  if (tid == 0) {
    float cv[G_GRP * TOPK_GROUP];
    int   ci[G_GRP * TOPK_GROUP];
    int nc = 0;
    for (int g = 0; g < G_GRP; ++g) {           // top-4 within each group of 8
      unsigned used = 0u;
      for (int k = 0; k < TOPK_GROUP; ++k) {
        int best = -1; float bv = -3.4e38f;
        for (int j = 0; j < 8; ++j) {
          if (!((used >> j) & 1u) && logits[g * 8 + j] > bv) { bv = logits[g * 8 + j]; best = j; }
        }
        used |= 1u << best;
        cv[nc] = bv; ci[nc] = g * 8 + best; ++nc;
      }
    }
    unsigned used2 = 0u;                        // top-8 of 32 candidates
    int   sel_i[TOPK]; float sel_v[TOPK];
    float sum = 0.f;
    for (int k = 0; k < TOPK; ++k) {
      int best = -1; float bv = -3.4e38f;
      for (int j = 0; j < G_GRP * TOPK_GROUP; ++j) {
        if (!((used2 >> j) & 1u) && cv[j] > bv) { bv = cv[j]; best = j; }
      }
      used2 |= 1u << best;
      sel_i[k] = ci[best]; sel_v[k] = bv; sum += bv;
    }
    const float inv = 1.f / (sum + 1e-20f);
    for (int k = 0; k < TOPK; ++k) {
      const int e = sel_i[k];
      const int pos = atomicAdd(&counts[e], 1);
      tok_list[e * T_TOK + pos] = t;
      wt_list[e * T_TOK + pos]  = sel_v[k] * inv;
    }
  }
}

// ---------------- WMMA fragment loaders ----------------
// A: 16x32 f16 per ISA layout (lanes 0-15: M=lane, K quads 0-7/16-23;
// lanes 16-31: same M, K quads 8-15/24-31). Per-lane halfs are contiguous,
// so these lower to 2x ds_load_b128.
__device__ __forceinline__ v16h load_a_lds(const _Float16* base, int ld, int k0) {
  const int lane = threadIdx.x & 31;
  const int m    = lane & 15;
  const int kh   = (lane & 16) ? 8 : 0;
  const _Float16* row = base + m * ld + k0 + kh;
  v16h a;
#pragma unroll
  for (int i = 0; i < 4; ++i) {
    a[2 * i]     = row[2 * i];
    a[2 * i + 1] = row[2 * i + 1];
    a[8 + 2 * i] = row[16 + 2 * i];
    a[9 + 2 * i] = row[17 + 2 * i];
  }
  return a;
}

// B: 32x16 f16 (K x N); lanes 0-15: N=lane, K=0..7/16-23; lanes 16-31: K=8..15/24-31.
// Source is row-major f32 (row stride ld), converted to f16 on the fly.
// Optionally issues a speculative global_prefetch for the next k-step
// (silently dropped by HW if the address is unmapped).
__device__ __forceinline__ v16h load_b_gmem(const float* base, int ld, int k0, int n0,
                                            bool prefetch_next) {
  const int lane = threadIdx.x & 31;
  const int n    = lane & 15;
  const int kh   = (lane & 16) ? 8 : 0;
  const float* p = base + (size_t)(k0 + kh) * ld + n0 + n;
  if (prefetch_next) __builtin_prefetch(p + (size_t)32 * ld, 0, 0);  // global_prefetch_b8
  v16h b;
#pragma unroll
  for (int v = 0; v < 8; ++v) {
    b[2 * v]     = (_Float16)p[(size_t)(2 * v + (v >= 4 ? 8 : 0)) * ld];
    b[2 * v + 1] = (_Float16)p[(size_t)(2 * v + 1 + (v >= 4 ? 8 : 0)) * ld];
  }
  return b;
}

// ---------------- fused FFN (gate_up -> silu*up -> down) ----------------
// One block = one expert x one 32-token tile. 16 waves (512 threads); each
// wave owns 64 output columns (4 N-tiles) and both 16-row M-tiles, so every
// B fragment feeds two WMMAs. LDS (dynamic, ~99.6 KB of the 320 KB WGP pool):
// x tile (f16, padded rows) aliased with the pre-activation tile, plus h tile.
#define BM        32
#define NTHREADS  512
#define XS_LD     1032                 // 1024 + 8 halfs pad -> distinct banks per row
#define HS_LD     520                  // 512 + 8
#define SM_A      (BM * XS_LD * 2)     // 66048 B (xs aliased with gu)
#define SM_H      (BM * HS_LD * 2)     // 33280 B
#define SM_TOTAL  (SM_A + SM_H + 256)

__global__ void __launch_bounds__(NTHREADS)
moe_ffn_kernel(const float* __restrict__ x,
               const float* __restrict__ w_gu_all,  // [ne][H][2I] row-major
               const float* __restrict__ w_d_all,   // [ne][I][H]  row-major
               const int*   __restrict__ counts,    // nullptr => shared-expert mode
               const int*   __restrict__ tok_list,
               const float* __restrict__ wt_list,
               float* __restrict__ out)
{
  extern __shared__ unsigned char smem[];
  _Float16* xs  = (_Float16*)smem;              // [BM][XS_LD]  GEMM1 A
  _Float16* gu  = (_Float16*)smem;              // [BM][XS_LD]  aliased (after barrier)
  _Float16* hs  = (_Float16*)(smem + SM_A);     // [BM][HS_LD]  GEMM2 A
  int*      tok = (int*)(smem + SM_A + SM_H);         // [BM]
  float*    twt = (float*)(smem + SM_A + SM_H + 128); // [BM]

  const int e    = blockIdx.y;
  const int tile = blockIdx.x;
  const int tid  = threadIdx.x;
  const int wave = tid >> 5;

  // token ids + per-token combine weights for this tile
  if (counts) {
    const int n = counts[e];
    if (tile * BM >= n) return;              // uniform across block
    if (tid < BM) {
      const int idx = tile * BM + tid;
      if (idx < n) { tok[tid] = tok_list[e * T_TOK + idx]; twt[tid] = wt_list[e * T_TOK + idx]; }
      else         { tok[tid] = tok_list[e * T_TOK + tile * BM]; twt[tid] = 0.f; }
    }
  } else {
    if (tid < BM) { tok[tid] = tile * BM + tid; twt[tid] = 1.f; }
  }
  __syncthreads();

  // stage gathered x rows into LDS as f16
  for (int i = tid; i < BM * H_DIM; i += NTHREADS) {
    const int m = i >> 10, c = i & (H_DIM - 1);
    xs[m * XS_LD + c] = (_Float16)x[(size_t)tok[m] * H_DIM + c];
  }
  __syncthreads();

  const float* wgu = w_gu_all + (size_t)e * H_DIM * (2 * I_DIM);
  const float* wd  = w_d_all  + (size_t)e * I_DIM * H_DIM;
  const int nbase  = wave * 64;               // 16 waves x 64 cols = 1024
  const v8f  vzero = {};

  // ---- GEMM1: pre_act[32, 2I] = xs[32,H] @ wgu[H, 2I] ----
  v8f acc[2][4];
#pragma unroll
  for (int mt = 0; mt < 2; ++mt)
#pragma unroll
    for (int nt = 0; nt < 4; ++nt) acc[mt][nt] = vzero;

  for (int k0 = 0; k0 < H_DIM; k0 += 32) {
    const bool pf = (k0 + 32) < H_DIM;
    const v16h a0 = load_a_lds(xs,               XS_LD, k0);
    const v16h a1 = load_a_lds(xs + 16 * XS_LD,  XS_LD, k0);
#pragma unroll
    for (int nt = 0; nt < 4; ++nt) {
      const v16h b = load_b_gmem(wgu, 2 * I_DIM, k0, nbase + nt * 16, pf);
      acc[0][nt] = __builtin_amdgcn_wmma_f32_16x16x32_f16(false, a0, false, b,
                                                          (short)0, acc[0][nt], false, false);
      acc[1][nt] = __builtin_amdgcn_wmma_f32_16x16x32_f16(false, a1, false, b,
                                                          (short)0, acc[1][nt], false, false);
    }
  }
  __syncthreads();   // all waves done reading xs; safe to overwrite alias
  {
    // C/D layout: lanes 0-15 -> M=r, lanes 16-31 -> M=r+8; N = lane%16
    const int lane = tid & 31;
    const int mrow = (lane & 16) ? 8 : 0;
    const int ncol = lane & 15;
#pragma unroll
    for (int mt = 0; mt < 2; ++mt)
#pragma unroll
      for (int nt = 0; nt < 4; ++nt)
#pragma unroll
        for (int r = 0; r < 8; ++r)
          gu[(mt * 16 + r + mrow) * XS_LD + nbase + nt * 16 + ncol] = (_Float16)acc[mt][nt][r];
  }
  __syncthreads();

  // ---- activation: hs = silu(gate) * up ----
  for (int i = tid; i < BM * I_DIM; i += NTHREADS) {
    const int m = i >> 9, j = i & (I_DIM - 1);
    const float g = (float)gu[m * XS_LD + j];
    const float u = (float)gu[m * XS_LD + I_DIM + j];
    const float s = g / (1.f + __expf(-g));
    hs[m * HS_LD + j] = (_Float16)(s * u);
  }
  __syncthreads();

  // ---- GEMM2: y[32,H] = hs[32,I] @ wd[I,H] ----
#pragma unroll
  for (int mt = 0; mt < 2; ++mt)
#pragma unroll
    for (int nt = 0; nt < 4; ++nt) acc[mt][nt] = vzero;

  for (int k0 = 0; k0 < I_DIM; k0 += 32) {
    const bool pf = (k0 + 32) < I_DIM;
    const v16h a0 = load_a_lds(hs,              HS_LD, k0);
    const v16h a1 = load_a_lds(hs + 16 * HS_LD, HS_LD, k0);
#pragma unroll
    for (int nt = 0; nt < 4; ++nt) {
      const v16h b = load_b_gmem(wd, H_DIM, k0, nbase + nt * 16, pf);
      acc[0][nt] = __builtin_amdgcn_wmma_f32_16x16x32_f16(false, a0, false, b,
                                                          (short)0, acc[0][nt], false, false);
      acc[1][nt] = __builtin_amdgcn_wmma_f32_16x16x32_f16(false, a1, false, b,
                                                          (short)0, acc[1][nt], false, false);
    }
  }

  // ---- scatter: out[token] += weight * y ----
  {
    const int lane = tid & 31;
    const int mrow = (lane & 16) ? 8 : 0;
    const int ncol = lane & 15;
#pragma unroll
    for (int mt = 0; mt < 2; ++mt)
#pragma unroll
      for (int r = 0; r < 8; ++r) {
        const int m = mt * 16 + r + mrow;
        const float w = twt[m];
        const size_t row = (size_t)tok[m] * H_DIM;
#pragma unroll
        for (int nt = 0; nt < 4; ++nt)
          atomicAdd(&out[row + nbase + nt * 16 + ncol], acc[mt][nt][r] * w);
      }
  }
}

// ---------------- host-side launch ----------------
extern "C" void kernel_launch(void* const* d_in, const int* in_sizes, int n_in,
                              void* d_out, int out_size, void* d_ws, size_t ws_size,
                              hipStream_t stream) {
  (void)in_sizes; (void)n_in; (void)out_size; (void)ws_size;
  const float* x      = (const float*)d_in[0];
  const float* gate_w = (const float*)d_in[1];
  const float* w_gu   = (const float*)d_in[2];
  const float* w_d    = (const float*)d_in[3];
  const float* s_gu   = (const float*)d_in[4];
  const float* s_d    = (const float*)d_in[5];
  float* out = (float*)d_out;

  // workspace: [counts: E ints pad->256B][tok_list: E*T ints][wt_list: E*T floats]
  int*   counts   = (int*)d_ws;
  int*   tok_list = (int*)((char*)d_ws + 256);
  float* wt_list  = (float*)((char*)d_ws + 256 + (size_t)E_EXP * T_TOK * sizeof(int));

  hipMemsetAsync(counts, 0, E_EXP * sizeof(int), stream);
  hipMemsetAsync(out, 0, (size_t)T_TOK * H_DIM * sizeof(float), stream);

  // 1) router
  moe_gate_kernel<<<T_TOK, 64, 0, stream>>>(x, gate_w, counts, tok_list, wt_list);

  // 2) shared experts (dense over all token tiles, weight 1.0)
  moe_ffn_kernel<<<dim3(T_TOK / BM, S_SH), NTHREADS, SM_TOTAL, stream>>>(
      x, s_gu, s_d, nullptr, nullptr, nullptr, out);

  // 3) routed experts (sparse: per-expert gathered token tiles)
  moe_ffn_kernel<<<dim3(T_TOK / BM, E_EXP), NTHREADS, SM_TOTAL, stream>>>(
      x, w_gu, w_d, counts, tok_list, wt_list, out);
}
